// SelfAttentionVanilla_46849503265234
// MI455X (gfx1250) — matrix operand order repaired and checked by reading, not compiled, not run on hardware
//
#include <hip/hip_runtime.h>
#include <cmath>

// Self-attention: B=2, N=2048, H=16, D=64, fp32 in/out.
// Flash-attention, f16 WMMA with f32 accumulation, wave32 / gfx1250.
// TRANSPOSED formulation: S^T = K*Q^T, O^T = V^T*P^T, so the softmax
// key-reduction is in-lane (one xor-16 shuffle per tile), stats are
// per-lane scalars, and P^T write-back is 4x ds_store_b128.
#define B_ 2
#define N_ 2048
#define H_ 16
#define D_ 64

typedef __attribute__((ext_vector_type(16))) _Float16 v16h;
typedef __attribute__((ext_vector_type(8)))  _Float16 v8h;
typedef __attribute__((ext_vector_type(8)))  float    v8f;
typedef __attribute__((ext_vector_type(4)))  float    f4;

__device__ __forceinline__ v16h cat8(v8h lo, v8h hi) {
  v16h r;
#pragma unroll
  for (int i = 0; i < 8; ++i) { r[i] = lo[i]; r[i + 8] = hi[i]; }
  return r;
}

#define WMMA_F16(A, B, C) \
  __builtin_amdgcn_wmma_f32_16x16x32_f16(false, (A), false, (B), (short)0, (C), false, false)

__global__ __launch_bounds__(128)
void fa_fwd_wmma(const float* __restrict__ Qp, const float* __restrict__ Kp,
                 const float* __restrict__ Vp, float* __restrict__ Op) {
  // K tile: 64 keys x 64 d (row-major), 72 halfs/row (144B rows, 16B-aligned)
  __shared__ __attribute__((aligned(32))) _Float16 ldsK[64][72];
  // V tile transposed: 64 d x 64 keys, padded to 72
  __shared__ __attribute__((aligned(32))) _Float16 ldsVt[64][72];
  // Per-wave P^T scratch stored query-major: 16 queries x 64 keys, padded to 72
  __shared__ __attribute__((aligned(32))) _Float16 ldsPq[4][16][72];

  const int bh   = blockIdx.y;
  const int b    = bh >> 4;        // / H_
  const int h    = bh & 15;        // % H_
  const int tid  = threadIdx.x;
  const int wave = tid >> 5;
  const int lane = tid & 31;
  const int g    = lane >> 4;      // lane half (wave32)
  const int ln   = lane & 15;
  const int q0   = blockIdx.x * 64 + wave * 16;   // this wave's 16 query rows

  const long RS = (long)H_ * D_;   // row stride in floats for [B,N,H,D]
  const float* qb = Qp + (long)b * N_ * RS + (long)h * D_;
  const float* kb = Kp + (long)b * N_ * RS + (long)h * D_;
  const float* vb = Vp + (long)b * N_ * RS + (long)h * D_;
  float*       ob = Op + (long)b * N_ * RS + (long)h * D_;

  // ---- Q^T as two f16 B-fragments (K=32 each); 1/sqrt(D) folded in ----
  // B layout (16-bit 32x16): lane col n=ln (query), elem i -> k(d) = c*32 + g*16 + i
  v16h bq[2];
  {
    const float* qr = qb + (long)(q0 + ln) * RS;
#pragma unroll
    for (int c = 0; c < 2; ++c) {
      const int d0 = c * 32 + g * 16;
#pragma unroll
      for (int run = 0; run < 4; ++run) {
        f4 x = *(const f4*)(qr + d0 + run * 4);
#pragma unroll
        for (int t = 0; t < 4; ++t)
          bq[c][run * 4 + t] = (_Float16)(x[t] * 0.125f);
      }
    }
  }

  // constant all-ones A fragment: ones(16x64) x P^T -> per-query key-sums
  v16h onesA;
#pragma unroll
  for (int i = 0; i < 16; ++i) onesA[i] = (_Float16)1.0f;

  v8f oacc[4] = {};                // O^T accumulator: 64d x 16q; elem r -> d = nb*16 + r + 8g
  float mq = -1.0e30f, lq = 0.0f;  // per-lane stats for query q0+ln

  const int srow = tid >> 1;          // staging: 2 threads per key row
  const int scol = (tid & 1) * 32;    // 32 contiguous d-columns each

  for (int j = 0; j < N_ / 64; ++j) {
    const int key0 = j * 64;
    __syncthreads();                   // previous tile's WMMA reads done
    {
      const float* kr = kb + (long)(key0 + srow) * RS + scol;
      const float* vr = vb + (long)(key0 + srow) * RS + scol;
#pragma unroll
      for (int i = 0; i < 32; i += 4) {
        f4 x = *(const f4*)(kr + i);
        f4 y = *(const f4*)(vr + i);
#pragma unroll
        for (int t = 0; t < 4; ++t) {
          ldsK[srow][scol + i + t]  = (_Float16)x[t];
          ldsVt[scol + i + t][srow] = (_Float16)y[t];   // transpose on the way in
        }
      }
      if (j + 1 < N_ / 64) {          // CDNA5 global_prefetch_b8 of next tile
        __builtin_prefetch(kr + 64 * RS, 0, 1);
        __builtin_prefetch(vr + 64 * RS, 0, 1);
      }
    }
    __syncthreads();

    // ---- S^T = K * (Q/sqrt(D))^T : four M-tiles of 16 keys ----
    // A layout (16-bit 16x32): lane row m=ln (key), elem i -> k=c*32+g*16+i
    v8f s[4] = {};
#pragma unroll
    for (int c = 0; c < 2; ++c) {
#pragma unroll
      for (int t = 0; t < 4; ++t) {
        v16h ak = cat8(*(const v8h*)&ldsK[t * 16 + ln][c * 32 + g * 16],
                       *(const v8h*)&ldsK[t * 16 + ln][c * 32 + g * 16 + 8]);
        s[t] = WMMA_F16(ak, bq[c], s[t]);
      }
    }

    // ---- per-lane softmax for query ln: max over 64 keys (in-lane + one xor16) ----
    float tmax = s[0][0];
#pragma unroll
    for (int t = 0; t < 4; ++t)
#pragma unroll
      for (int r = 0; r < 8; ++r) tmax = fmaxf(tmax, s[t][r]);
    tmax = fmaxf(tmax, __shfl_xor(tmax, 16, 32));   // merge other lane-half's 32 keys
    const float mn = fmaxf(mq, tmax);
    const float al = __expf(mq - mn);
    mq = mn;
    lq *= al;
#pragma unroll
    for (int nb = 0; nb < 4; ++nb)
#pragma unroll
      for (int r = 0; r < 8; ++r) oacc[nb][r] *= al;

    // ---- exp -> P^T stored query-major: row ln, keys t*16 + 8g + (0..7) ----
#pragma unroll
    for (int t = 0; t < 4; ++t) {
      v8h ph;
#pragma unroll
      for (int r = 0; r < 8; ++r) ph[r] = (_Float16)__expf(s[t][r] - mn);
      *(v8h*)&ldsPq[wave][ln][t * 16 + 8 * g] = ph;   // one ds_store_b128
    }

    // ---- P^T as two B fragments (key chunks of 32); same wave: DS in-order ----
    v16h bp0 = cat8(*(const v8h*)&ldsPq[wave][ln][g * 16],
                    *(const v8h*)&ldsPq[wave][ln][g * 16 + 8]);
    v16h bp1 = cat8(*(const v8h*)&ldsPq[wave][ln][32 + g * 16],
                    *(const v8h*)&ldsPq[wave][ln][32 + g * 16 + 8]);

    // ---- key-sums of P via ones-WMMA (replicated across rows; take elem 0) ----
    v8f psum = {};
    psum = WMMA_F16(onesA, bp0, psum);
    psum = WMMA_F16(onesA, bp1, psum);
    lq += psum[0];

    // ---- O^T += V^T(64d x 64k) * P^T(64k x 16q) ----
#pragma unroll
    for (int c = 0; c < 2; ++c) {
#pragma unroll
      for (int nb = 0; nb < 4; ++nb) {
        v16h av = cat8(*(const v8h*)&ldsVt[nb * 16 + ln][c * 32 + g * 16],
                       *(const v8h*)&ldsVt[nb * 16 + ln][c * 32 + g * 16 + 8]);
        oacc[nb] = WMMA_F16(av, c ? bp1 : bp0, oacc[nb]);
      }
    }
  }

  // ---- epilogue: O /= key-sum; 8 consecutive d per v8f -> vectorized stores ----
  {
    const float inv = 1.0f / lq;
    float* orow = ob + (long)(q0 + ln) * RS;
#pragma unroll
    for (int nb = 0; nb < 4; ++nb) {
      f4 lo, hi;
#pragma unroll
      for (int t = 0; t < 4; ++t) { lo[t] = oacc[nb][t] * inv; hi[t] = oacc[nb][4 + t] * inv; }
      *(f4*)(orow + nb * 16 + 8 * g)     = lo;
      *(f4*)(orow + nb * 16 + 8 * g + 4) = hi;
    }
  }
}

extern "C" void kernel_launch(void* const* d_in, const int* in_sizes, int n_in,
                              void* d_out, int out_size, void* d_ws, size_t ws_size,
                              hipStream_t stream) {
  (void)in_sizes; (void)n_in; (void)out_size; (void)d_ws; (void)ws_size;
  const float* q = (const float*)d_in[0];
  const float* k = (const float*)d_in[1];
  const float* v = (const float*)d_in[2];
  float* o = (float*)d_out;
  dim3 grid(N_ / 64, B_ * H_);   // 32 query tiles x 32 (b,h) heads = 1024 blocks
  fa_fwd_wmma<<<grid, 128, 0, stream>>>(q, k, v, o);
}
